// VQAutoEncoder_55722905699296
// MI455X (gfx1250) — matrix operand order repaired and checked by reading, not compiled, not run on hardware
//
#include <hip/hip_runtime.h>
#include <cmath>

// ---------------------------------------------------------------------------
// CDNA5 (gfx1250) VQ-VAE forward.  Channels-last activations, bf16-repacked
// weights, LDS-staged implicit-GEMM conv3d on v_wmma_f32_16x16x32_bf16,
// TDM (tensor_load_to_lds) weight tiles, branchless vectorized fills, wave32.
// ---------------------------------------------------------------------------

typedef __bf16 bf16;
typedef __attribute__((ext_vector_type(16))) __bf16 v16bf;
typedef __attribute__((ext_vector_type(8)))  __bf16 v8bf;
typedef __attribute__((ext_vector_type(4)))  __bf16 v4bf;
typedef __attribute__((ext_vector_type(8)))  float  v8f;

#if defined(__has_builtin)
# if __has_builtin(__builtin_amdgcn_tensor_load_to_lds) && \
     __has_builtin(__builtin_amdgcn_s_wait_tensorcnt)
#  define HAVE_TDM 1
# endif
#endif
#ifndef HAVE_TDM
# define HAVE_TDM 0
#endif

#if HAVE_TDM
typedef __attribute__((ext_vector_type(4))) unsigned v4u;
typedef __attribute__((ext_vector_type(8))) int      v8i;
typedef __attribute__((ext_vector_type(4))) int      v4i;
#endif

__device__ __forceinline__ bf16 to_bf16(float f) { return (bf16)f; }

__device__ __forceinline__ v8bf cvt8(const float4& a, const float4& b) {
  v8bf t;
  t[0] = (bf16)a.x; t[1] = (bf16)a.y; t[2] = (bf16)a.z; t[3] = (bf16)a.w;
  t[4] = (bf16)b.x; t[5] = (bf16)b.y; t[6] = (bf16)b.z; t[7] = (bf16)b.w;
  return t;
}

// ---------------------------------------------------------------------------
// NCDHW -> channels-last ([b][voxel][c]) transpose.
// ---------------------------------------------------------------------------
__global__ void to_cl_k(const float* __restrict__ in, float* __restrict__ out,
                        int C, long S, long total)
{
  const long t = (long)blockIdx.x * blockDim.x + threadIdx.x;
  if (t >= total) return;
  const long b = t / (C * S);
  const long rem = t - b * C * S;
  const int  c = (int)(rem / S);
  const long s = rem - (long)c * S;
  out[(b * S + s) * C + c] = in[t];
}

// Weight repack: [cout][cin][k^3] f32 -> [cout][k^3][cin] bf16.
__global__ void repack_w_k(const float* __restrict__ w, bf16* __restrict__ wp,
                           int Cin, int K3, long total)
{
  const long t = (long)blockIdx.x * blockDim.x + threadIdx.x;
  if (t >= total) return;
  const long co = t / ((long)Cin * K3);
  const long rem = t - co * Cin * K3;
  const int ci = (int)(rem / K3);
  const int tap = (int)(rem - (long)ci * K3);
  wp[(co * K3 + tap) * Cin + ci] = to_bf16(w[t]);
}

// ---------------------------------------------------------------------------
// Implicit-GEMM conv3d, channels-last.  Block = 256 threads = 8 waves,
// tile 64(M=voxels) x 32(N=cout).  K iterated as (tap, 32-channel chunk).
// Channel counts are multiples of 8, so fills are branchless: clamped vector
// load + cndmask zero.  LDS A/B rows kept in natural k order so each lane's WMMA
// fragment is one contiguous 32B ds load.  B tiles come via the TDM.
// ---------------------------------------------------------------------------
#define BM 64
#define BN 32

__global__ __launch_bounds__(256) void conv3d_cl_wmma(
    const float* __restrict__ X,     // [B][Rin^3][Cin]  channels-last
    const bf16*  __restrict__ Wp,    // [Cout][K3][Cin]  bf16 packed
    const float* __restrict__ Bias,  // [Cout]
    const float* __restrict__ Res,   // [B][Rout^3][Cout] cl, or null
    float* __restrict__ Y,
    int B, int Cin, int Rin, int Cout, int Ksz, int Rout,
    int stride, int pad, int upsample, int out_cf)
{
  __shared__ __align__(64) bf16 As[BM][32];   // natural k order per row
  __shared__ __align__(64) bf16 Bs[BN][32];   // natural k order per row

  const int tid = threadIdx.x;
  const int lane = tid & 31;
  const int wave = tid >> 5;
  const int half = lane >> 4, low = lane & 15;

  const int Rout3 = Rout * Rout * Rout;
  const int Rin3 = Rin * Rin * Rin;
  const int M = B * Rout3;
  const int m_blk = blockIdx.x * BM;
  const int n_blk = blockIdx.y * BN;
  const int K2 = Ksz * Ksz, K3 = K2 * Ksz;
  const int RinU = upsample ? (Rin << 1) : Rin;

  // A-fill assignment: row constant across all chunks.
  const int arow = tid & (BM - 1);
  const int akg = tid >> 6;                                  // k-group of 8
  const int aphys = akg << 3;                                // 0,8,16,24
  const int am = m_blk + arow;
  const bool amv = am < M;
  int cm = amv ? am : 0;
  const int aow = cm % Rout; cm /= Rout;
  const int aoh = cm % Rout; cm /= Rout;
  const int aod = cm % Rout; cm /= Rout;
  const int abt = cm;
  // B-fill assignment.
  const int brow = tid & (BN - 1);
  const int bk4 = (tid >> 5) * 4;
  const int bco = n_blk + brow;
  const bool bcv = bco < Cout;

  // Per-wave LDS fragment pointers (constant).
  const int wM = wave & 3;
  const int wN = wave >> 2;
  const bf16* aptr = &As[wM * 16 + low][half * 16];
  const bf16* bptr = &Bs[wN * 16 + low][half * 16];

  v8f acc = {};

  for (int tap = 0; tap < K3; ++tap) {
    const int kd = tap / K2;
    const int kh = (tap / Ksz) % Ksz;
    const int kw = tap % Ksz;
    int id = aod * stride - pad + kd;
    int ih = aoh * stride - pad + kh;
    int iw = aow * stride - pad + kw;
    const bool tv = amv && id >= 0 && id < RinU && ih >= 0 && ih < RinU &&
                    iw >= 0 && iw < RinU;
    if (upsample) { id >>= 1; ih >>= 1; iw >>= 1; }
    const long vbase = tv ? (((long)abt * Rin3 + ((long)id * Rin + ih) * Rin + iw)
                            * (long)Cin) : 0;
    const long wbase = ((long)bco * K3 + tap) * (long)Cin;
    if (bcv) __builtin_prefetch((const void*)(Wp + wbase), 0, 0);

    for (int cb = 0; cb < Cin; cb += 32) {
      // ---- A tile fill: branchless (Cin is a multiple of 8) ----
      {
        const int c0 = cb + akg * 8;
        const bool gv = tv && (c0 + 8 <= Cin);
        const long off = gv ? (vbase + c0) : 0;
        float4 f0 = *(const float4*)(X + off);
        float4 f1 = *(const float4*)(X + off + 4);
        if (!gv) { f0 = make_float4(0.f, 0.f, 0.f, 0.f);
                   f1 = make_float4(0.f, 0.f, 0.f, 0.f); }
        *(v8bf*)(&As[arow][aphys]) = cvt8(f0, f1);
      }
      // ---- B tile: TDM when full, branchless vector fill otherwise ----
      bool used_tdm = false;
#if HAVE_TDM
      if (cb + 32 <= Cin && n_blk + 32 <= Cout) {
        used_tdm = true;
        if (wave == 0) {
          const unsigned ldsa = (unsigned)(unsigned long long)(&Bs[0][0]);
          const unsigned long long ga =
              (unsigned long long)(const void*)(Wp + ((long)n_blk * K3 + tap) * Cin + cb);
          v4u g0 = { 1u,                      // count=1 (valid user D#)
                     ldsa,                    // lds_addr
                     (unsigned)ga,            // global_addr[31:0]
                     (unsigned)((ga >> 32) & 0x01FFFFFFu) | (2u << 30) }; // type=2
          const unsigned d0 = 1u << 20;      // tensor_dim0 (large, no OOB)
          const unsigned d1 = 1u << 20;      // tensor_dim1
          const unsigned long long st0 = (unsigned long long)K3 * (unsigned)Cin;
          v8i g1 = { (int)0x00010000u,                         // data_size=2B
                     (int)((d0 & 0xFFFFu) << 16),              // dim0 lo
                     (int)((d0 >> 16) | ((d1 & 0xFFFFu) << 16)),
                     (int)((d1 >> 16) | (32u << 16)),          // tile_dim0=32
                     (int)32u,                                 // tile_dim1=32
                     (int)(unsigned)st0,                       // dim0_stride lo
                     (int)(unsigned)(st0 >> 32),               // dim0_stride hi
                     0 };
          v4i gz = { 0, 0, 0, 0 };
          v8i gz8 = { 0, 0, 0, 0, 0, 0, 0, 0 };
          __builtin_amdgcn_tensor_load_to_lds(g0, g1, gz, gz, gz8, 0);
          __builtin_amdgcn_s_wait_tensorcnt(0);
        }
      }
#endif
      if (!used_tdm) {
        const int c0 = cb + bk4;
        const bool gv = bcv && (c0 + 4 <= Cin);   // Cin multiple of 4
        const long off = gv ? (wbase + c0) : 0;
        v4bf t = *(const v4bf*)(Wp + off);
        const v4bf z4 = {};
        if (!gv) t = z4;
        *(v4bf*)(&Bs[brow][bk4]) = t;
      }
      __syncthreads();
      // ---- compute: one contiguous 32B LDS load per fragment + WMMA ----
      const v16bf a = *(const v16bf*)aptr;
      const v16bf b = *(const v16bf*)bptr;
      acc = __builtin_amdgcn_wmma_f32_16x16x32_bf16(false, a, false, b, (short)0,
                                                    acc, false, false);
      __syncthreads();
    }
  }

  // ---- store (channels-last, or channels-first for V-proj / final layer) ----
  const int n = n_blk + wN * 16 + low;
  if (n >= Cout) return;
  const float bias = Bias ? Bias[n] : 0.f;
#pragma unroll
  for (int r = 0; r < 8; ++r) {
    const int m = m_blk + wM * 16 + r + 8 * half;
    if (m >= M) continue;
    float val = acc[r] + bias;
    if (Res) val += Res[(long)m * Cout + n];
    if (out_cf) {
      const int ob = m / Rout3, vox = m - ob * Rout3;
      Y[((long)ob * Cout + n) * Rout3 + vox] = val;
    } else {
      Y[(long)m * Cout + n] = val;
    }
  }
}

// ---------------------------------------------------------------------------
// Batched WMMA GEMM with K-contiguous operands (all call sites):
//   C[z,m,n] = alpha * sum_k A[z, m*as_m + k] * B[z, n*bs_n + k]
// K is a multiple of 32; fragments are straight float4 loads + cvt.
// ---------------------------------------------------------------------------
__global__ __launch_bounds__(128) void gemm_wmma(
    const float* __restrict__ A, const float* __restrict__ Bm,
    float* __restrict__ C,
    int M, int N, int K,
    long abat, long as_m,
    long bbat, long bs_n,
    long cbat, long cs_m, long cs_n,
    float alpha)
{
  const int lane = threadIdx.x & 31;
  const int wave = threadIdx.x >> 5;
  const int z = blockIdx.z;
  const int m0 = (blockIdx.x * 4 + wave) * 16;
  const int n0 = blockIdx.y * 16;
  if (m0 >= M) return;
  const int half = lane >> 4, low = lane & 15;

  const int am = m0 + low;
  const bool aValid = am < M;
  const int bn = n0 + low;
  const bool bValid = bn < N;
  const float* Ab = A + (long)z * abat + (aValid ? (long)am * as_m : 0);
  const float* Bb = Bm + (long)z * bbat + (bValid ? (long)bn * bs_n : 0);

  v8f acc = {};
  for (int k0 = 0; k0 < K; k0 += 32) {
    // A fragment: k = {half*8..+8, 16+half*8..+8}  (matches 16-bit A layout)
    const float* ap = Ab + k0 + half * 8;
    const float4 a0 = *(const float4*)(ap);
    const float4 a1 = *(const float4*)(ap + 4);
    const float4 a2 = *(const float4*)(ap + 16);
    const float4 a3 = *(const float4*)(ap + 20);
    const v16bf a = __builtin_shufflevector(cvt8(a0, a1), cvt8(a2, a3),
        0, 1, 2, 3, 4, 5, 6, 7, 8, 9, 10, 11, 12, 13, 14, 15);
    // B fragment: k = {half*8..+8, 16+half*8..+8}  (same order as A)
    const float* bp = Bb + k0 + half * 8;
    const float4 b0 = *(const float4*)(bp);
    const float4 b1 = *(const float4*)(bp + 4);
    const float4 b2 = *(const float4*)(bp + 16);
    const float4 b3 = *(const float4*)(bp + 20);
    const v16bf b = __builtin_shufflevector(cvt8(b0, b1), cvt8(b2, b3),
        0, 1, 2, 3, 4, 5, 6, 7, 8, 9, 10, 11, 12, 13, 14, 15);
    acc = __builtin_amdgcn_wmma_f32_16x16x32_bf16(false, a, false, b, (short)0,
                                                  acc, false, false);
  }
  const int n = n0 + low;
  if (n >= N) return;
#pragma unroll
  for (int r = 0; r < 8; ++r) {
    const int m = m0 + r + 8 * half;
    if (m >= M) continue;
    C[(long)z * cbat + (long)m * cs_m + (long)n * cs_n] = alpha * acc[r];
  }
}

// ---------------------------------------------------------------------------
// GroupNorm (+ optional swish), channels-last.  One block per (batch, group).
// ---------------------------------------------------------------------------
__global__ __launch_bounds__(256) void group_norm_cl_k(
    const float* __restrict__ X, const float* __restrict__ Gw,
    const float* __restrict__ Bw, float* __restrict__ Y,
    int C, long S, int groups, int do_swish)
{
  __shared__ float s_sum[256];
  __shared__ float s_sq[256];
  const int tid = threadIdx.x;
  const int grp = blockIdx.x % groups;
  const int bb = blockIdx.x / groups;
  const int cg = C / groups;
  const int cgs = 31 - __clz(cg);          // cg is a power of two (2/4/8)
  const long cnt = S << cgs;
  const long xb = (long)bb * S * C + (long)grp * cg;

  float s = 0.f, ss = 0.f;
  for (long i = tid; i < cnt; i += 256) {
    const long sp = i >> cgs;
    const int j = (int)i & (cg - 1);
    const float v = X[xb + sp * C + j];
    s += v; ss += v * v;
  }
  s_sum[tid] = s; s_sq[tid] = ss;
  __syncthreads();
  for (int off = 128; off > 0; off >>= 1) {
    if (tid < off) { s_sum[tid] += s_sum[tid + off]; s_sq[tid] += s_sq[tid + off]; }
    __syncthreads();
  }
  const float mean = s_sum[0] / (float)cnt;
  const float var = s_sq[0] / (float)cnt - mean * mean;
  const float rstd = rsqrtf(var + 1e-6f);
  for (long i = tid; i < cnt; i += 256) {
    const long sp = i >> cgs;
    const int j = (int)i & (cg - 1);
    const int c = grp * cg + j;
    float v = (X[xb + sp * C + j] - mean) * rstd * Gw[c] + Bw[c];
    if (do_swish) v = v / (1.f + __expf(-v));
    Y[xb + sp * C + j] = v;
  }
}

// Row softmax, in place.
__global__ __launch_bounds__(256) void softmax_rows_k(float* __restrict__ W, int n)
{
  __shared__ float red[256];
  const int tid = threadIdx.x;
  float* row = W + (long)blockIdx.x * n;
  float mx = -3.4e38f;
  for (int i = tid; i < n; i += 256) mx = fmaxf(mx, row[i]);
  red[tid] = mx; __syncthreads();
  for (int off = 128; off > 0; off >>= 1) {
    if (tid < off) red[tid] = fmaxf(red[tid], red[tid + off]);
    __syncthreads();
  }
  const float m = red[0];
  __syncthreads();
  float s = 0.f;
  for (int i = tid; i < n; i += 256) { float e = __expf(row[i] - m); row[i] = e; s += e; }
  red[tid] = s; __syncthreads();
  for (int off = 128; off > 0; off >>= 1) {
    if (tid < off) red[tid] += red[tid + off];
    __syncthreads();
  }
  const float inv = 1.f / red[0];
  for (int i = tid; i < n; i += 256) row[i] *= inv;
}

// Codebook row L2^2 norms.
__global__ void row_norm2_k(const float* __restrict__ E, float* __restrict__ out,
                            int rows, int cols)
{
  const int r = blockIdx.x * blockDim.x + threadIdx.x;
  if (r >= rows) return;
  const float* p = E + (long)r * cols;
  float s = 0.f;
  for (int c = 0; c < cols; ++c) s += p[c] * p[c];
  out[r] = s;
}

// argmin_e ( ||e||^2 - 2*dot[r][e] ), first-index tiebreak.
__global__ __launch_bounds__(256) void argmin_rows_k(
    const float* __restrict__ dot, const float* __restrict__ en,
    int* __restrict__ idx, int n)
{
  __shared__ float sv[256];
  __shared__ int   si[256];
  const int tid = threadIdx.x;
  const float* row = dot + (long)blockIdx.x * n;
  float bv = 3.4e38f; int bi = 0;
  for (int e = tid; e < n; e += 256) {
    const float v = en[e] - 2.f * row[e];
    if (v < bv || (v == bv && e < bi)) { bv = v; bi = e; }
  }
  sv[tid] = bv; si[tid] = bi; __syncthreads();
  for (int off = 128; off > 0; off >>= 1) {
    if (tid < off) {
      if (sv[tid + off] < sv[tid] ||
          (sv[tid + off] == sv[tid] && si[tid + off] < si[tid])) {
        sv[tid] = sv[tid + off]; si[tid] = si[tid + off];
      }
    }
    __syncthreads();
  }
  if (tid == 0) idx[blockIdx.x] = si[0];
}

// zq (channels-last) [b][s][c] = emb[idx[b*S+s]][c]  (contiguous writes+reads)
__global__ void gather_zq_k(const float* __restrict__ E, const int* __restrict__ idx,
                            float* __restrict__ zq, int C, int S, long total)
{
  const long t = (long)blockIdx.x * blockDim.x + threadIdx.x;
  if (t >= total) return;
  const long b = t / ((long)S * C);
  const long rem = t - b * S * C;
  const int s = (int)(rem / C);
  const int c = (int)(rem - (long)s * C);
  zq[t] = E[(long)idx[b * S + s] * C + c];
}

// ---------------------------------------------------------------------------
// Host-side plan walker.
// ---------------------------------------------------------------------------
namespace {

constexpr int BATCH = 2, IN_CH = 8, NF = 64, NUM_RES = 2, RES = 32;
constexpr int EMB = 256, CODEBOOK = 1024, GROUPS = 32;
constexpr int CH_MULT[3] = {1, 2, 4};

struct Cursor { void* const* in; int i;
  const float* next() { return (const float*)in[i++]; } };
struct ConvP { const float* b; const float* w; };
struct NormP { const float* b; const float* g; };
struct ResP  { ConvP c1, c2, co; bool has_co; NormP n1, n2; };
struct AttnP { ConvP k; NormP n; ConvP proj, q, v; };

// jax tree-flatten order: dict keys sorted alphabetically.
ConvP rd_conv(Cursor& c){ ConvP p; p.b=c.next(); p.w=c.next(); return p; }
NormP rd_norm(Cursor& c){ NormP p; p.b=c.next(); p.g=c.next(); return p; }
ResP  rd_res (Cursor& c, bool co){ ResP r; r.c1=rd_conv(c); r.c2=rd_conv(c);
  r.has_co=co; if(co) r.co=rd_conv(c); r.n1=rd_norm(c); r.n2=rd_norm(c); return r; }
AttnP rd_attn(Cursor& c){ AttnP a; a.k=rd_conv(c); a.n=rd_norm(c);
  a.proj=rd_conv(c); a.q=rd_conv(c); a.v=rd_conv(c); return a; }

inline int ceil_div(long a, long b){ return (int)((a + b - 1) / b); }

struct RunState {
  const float* cur;
  float* o0; float* o1; float* tmp;
  bf16* wp;
  hipStream_t st;
};
inline void advance(RunState& rs) {
  rs.cur = rs.o0; float* t = rs.o0; rs.o0 = rs.o1; rs.o1 = t;
}

void conv_launch(RunState& rs, const float* X, const ConvP& p, const float* Res,
                 float* Y, int Cin, int Rin, int Cout, int Ksz, int Rout,
                 int stride, int pad, int up, int out_cf)
{
  const int K3 = Ksz * Ksz * Ksz;
  const long wtot = (long)Cout * Cin * K3;
  repack_w_k<<<ceil_div(wtot, 256), 256, 0, rs.st>>>(p.w, rs.wp, Cin, K3, wtot);
  const int M = BATCH * Rout * Rout * Rout;
  dim3 grid(ceil_div(M, BM), ceil_div(Cout, BN), 1);
  conv3d_cl_wmma<<<grid, 256, 0, rs.st>>>(X, rs.wp, p.b, Res, Y, BATCH, Cin, Rin,
                                          Cout, Ksz, Rout, stride, pad, up, out_cf);
}

void gn_launch(RunState& rs, const float* X, const NormP& p, float* Y,
               int C, int R, int swish)
{
  group_norm_cl_k<<<BATCH * GROUPS, 256, 0, rs.st>>>(X, p.g, p.b, Y, C,
                                                     (long)R * R * R, GROUPS, swish);
}

void run_res(RunState& rs, const ResP& p, int Cin, int Cout, int R)
{
  gn_launch(rs, rs.cur, p.n1, rs.tmp, Cin, R, 1);
  conv_launch(rs, rs.tmp, p.c1, nullptr, rs.o0, Cin, R, Cout, 3, R, 1, 1, 0, 0);
  gn_launch(rs, rs.o0, p.n2, rs.tmp, Cout, R, 1);
  if (p.has_co) {
    conv_launch(rs, rs.cur, p.co, nullptr, rs.o0, Cin, R, Cout, 1, R, 1, 0, 0, 0);
    conv_launch(rs, rs.tmp, p.c2, rs.o0, rs.o0, Cout, R, Cout, 3, R, 1, 1, 0, 0);
  } else {
    conv_launch(rs, rs.tmp, p.c2, rs.cur, rs.o0, Cout, R, Cout, 3, R, 1, 1, 0, 0);
  }
  advance(rs);
}

void run_attn(RunState& rs, const AttnP& p, int C, int R,
              float* QB, float* KB, float* VB, float* WB)
{
  const int S = R * R * R;
  gn_launch(rs, rs.cur, p.n, rs.tmp, C, R, 0);
  conv_launch(rs, rs.tmp, p.q, nullptr, QB, C, R, C, 1, R, 1, 0, 0, 0);
  conv_launch(rs, rs.tmp, p.k, nullptr, KB, C, R, C, 1, R, 1, 0, 0, 0);
  // V written channels-first ([b][c][s]) so softmax*V's B operand is K-contig.
  conv_launch(rs, rs.tmp, p.v, nullptr, VB, C, R, C, 1, R, 1, 0, 0, 1);
  const float alpha = 1.f / sqrtf((float)C);
  {
    // scores[b][i][j] = alpha * sum_c q[b][i][c] * k[b][j][c]
    dim3 g(ceil_div(ceil_div(S, 16), 4), ceil_div(S, 16), BATCH);
    gemm_wmma<<<g, 128, 0, rs.st>>>(QB, KB, WB, S, S, C,
        (long)S * C, C,
        (long)S * C, C,
        (long)S * S, S, 1, alpha);
  }
  softmax_rows_k<<<BATCH * S, 256, 0, rs.st>>>(WB, S);
  {
    // h[b][i][c] = sum_j w[b][i][j] * v[b][c][j]   (V is channels-first)
    dim3 g(ceil_div(ceil_div(S, 16), 4), ceil_div(C, 16), BATCH);
    gemm_wmma<<<g, 128, 0, rs.st>>>(WB, VB, rs.tmp, S, C, S,
        (long)S * S, S,
        (long)C * S, S,
        (long)S * C, C, 1, 1.0f);
  }
  conv_launch(rs, rs.tmp, p.proj, rs.cur, rs.o0, C, R, C, 1, R, 1, 0, 0, 0);
  advance(rs);
}

} // namespace

extern "C" void kernel_launch(void* const* d_in, const int* in_sizes, int n_in,
                              void* d_out, int out_size, void* d_ws, size_t ws_size,
                              hipStream_t stream)
{
  (void)in_sizes; (void)n_in; (void)out_size; (void)ws_size;

  // Input order assumption: setup_inputs() insertion order (x, enc_params,
  // embedding, dec_params); nested dicts flattened in sorted-key (jax) order.
  Cursor pc{d_in, 0};
  const float* x = pc.next();                 // [2,8,32,32,32] channels-first

  float* ws = (float*)d_ws;
  const long ACT = 8L * 1024 * 1024 + 256;    // >= 2*128*32^3 floats
  float* A0 = ws;
  float* A1 = ws + ACT;
  float* A2 = ws + 2 * ACT;
  float* QB = ws + 3 * ACT;                   // 2*512*256
  float* KB = QB + 262144;
  float* VB = KB + 262144;
  float* WB = VB + 262144;                    // 2*512*512
  float* QDOT = WB + 524288;                  // 2*512*1024
  float* ENORM = QDOT + 1048576;              // 1024
  int*   IDX = (int*)(ENORM + 1024);          // 1024
  bf16*  WP  = (bf16*)(IDX + 1024);           // <= 256*256*27 bf16 (reused)

  // Transpose input to channels-last.
  {
    const long total = (long)BATCH * IN_CH * RES * RES * RES;
    to_cl_k<<<ceil_div(total, 256), 256, 0, stream>>>(x, A0, IN_CH,
                                                      (long)RES * RES * RES, total);
  }

  RunState rs{A0, A1, A0, A2, WP, stream};

  // ---------------- encoder ----------------
  int R = RES;
  { ConvP p = rd_conv(pc);
    conv_launch(rs, rs.cur, p, nullptr, rs.o0, IN_CH, R, NF, 3, R, 1, 1, 0, 0);
    advance(rs); }
  const int in_mult[3] = {1, CH_MULT[0], CH_MULT[1]};
  int C = NF;
  for (int i = 0; i < 3; ++i) {
    int cin = NF * in_mult[i];
    const int cout = NF * CH_MULT[i];
    for (int j = 0; j < NUM_RES; ++j) {
      ResP rp = rd_res(pc, cin != cout);
      run_res(rs, rp, cin, cout, R);
      cin = cout;
      if (R == 8) { AttnP ap = rd_attn(pc); run_attn(rs, ap, cout, R, QB, KB, VB, WB); }
    }
    C = cout;
    if (i != 2) {
      ConvP p = rd_conv(pc);   // downsample: asym end-pad via bounds check
      conv_launch(rs, rs.cur, p, nullptr, rs.o0, C, R, C, 3, R / 2, 2, 0, 0, 0);
      advance(rs); R /= 2;
    }
  }
  { ResP rp = rd_res(pc, false); run_res(rs, rp, C, C, R); }
  { AttnP ap = rd_attn(pc); run_attn(rs, ap, C, R, QB, KB, VB, WB); }
  { ResP rp = rd_res(pc, false); run_res(rs, rp, C, C, R); }
  { NormP np = rd_norm(pc);
    gn_launch(rs, rs.cur, np, rs.o0, C, R, 0); advance(rs); }
  { ConvP p = rd_conv(pc);
    conv_launch(rs, rs.cur, p, nullptr, rs.o0, C, R, EMB, 3, R, 1, 1, 0, 0);
    advance(rs); C = EMB; }

  // ---------------- quantize ----------------
  const float* emb = pc.next();               // [1024, 256]
  {
    const int S = R * R * R;                  // 512
    row_norm2_k<<<ceil_div(CODEBOOK, 256), 256, 0, stream>>>(emb, ENORM, CODEBOOK, EMB);
    // dot[b][s][e] = sum_c z[b][s][c] * emb[e][c]   (both K-contiguous)
    dim3 g(ceil_div(ceil_div(S, 16), 4), ceil_div(CODEBOOK, 16), BATCH);
    gemm_wmma<<<g, 128, 0, stream>>>(rs.cur, emb, QDOT, S, CODEBOOK, EMB,
        (long)S * EMB, EMB,
        0, EMB,
        (long)S * CODEBOOK, CODEBOOK, 1, 1.0f);
    argmin_rows_k<<<BATCH * S, 256, 0, stream>>>(QDOT, ENORM, IDX, CODEBOOK);
    const long total = (long)BATCH * S * EMB;
    gather_zq_k<<<ceil_div(total, 256), 256, 0, stream>>>(emb, IDX, rs.o0,
                                                          EMB, S, total);
    advance(rs);
  }

  // ---------------- decoder ----------------
  C = NF * CH_MULT[2];                        // 256
  { ConvP p = rd_conv(pc);
    conv_launch(rs, rs.cur, p, nullptr, rs.o0, EMB, R, C, 3, R, 1, 1, 0, 0);
    advance(rs); }
  { ResP rp = rd_res(pc, false); run_res(rs, rp, C, C, R); }
  { AttnP ap = rd_attn(pc); run_attn(rs, ap, C, R, QB, KB, VB, WB); }
  { ResP rp = rd_res(pc, false); run_res(rs, rp, C, C, R); }
  int cin = C;
  for (int i = 2; i >= 0; --i) {
    const int cout = NF * CH_MULT[i];
    for (int j = 0; j < NUM_RES; ++j) {
      ResP rp = rd_res(pc, cin != cout);
      run_res(rs, rp, cin, cout, R);
      cin = cout;
      if (R == 8) { AttnP ap = rd_attn(pc); run_attn(rs, ap, cout, R, QB, KB, VB, WB); }
    }
    if (i != 0) {
      ConvP p = rd_conv(pc);                  // nearest-2x upsample fused in conv
      conv_launch(rs, rs.cur, p, nullptr, rs.o0, cin, R, cin, 3, R * 2, 1, 1, 1, 0);
      advance(rs); R *= 2;
    }
  }
  C = cin;                                    // 64
  { NormP np = rd_norm(pc);
    gn_launch(rs, rs.cur, np, rs.o0, C, R, 0); advance(rs); }
  { ConvP p = rd_conv(pc);                    // final conv -> channels-first d_out
    conv_launch(rs, rs.cur, p, nullptr, (float*)d_out, C, R, IN_CH, 3, R, 1, 1, 0, 1); }
}
